// SwinTransformer_31379031064912
// MI455X (gfx1250) — matrix operand order repaired and checked by reading, not compiled, not run on hardware
//
#include <hip/hip_runtime.h>
#include <hip/hip_bf16.h>
#include <math.h>

typedef _Float16 h16;
typedef __attribute__((ext_vector_type(16))) _Float16 v16h;
typedef __attribute__((ext_vector_type(8)))  _Float16 v8h;
typedef __attribute__((ext_vector_type(8)))  float    v8f;
typedef __attribute__((ext_vector_type(4)))  unsigned int v4u;
typedef __attribute__((ext_vector_type(8)))  unsigned int v8u;

#define BB     32
#define HH     56
#define DIMC   512
#define WSZ    7
#define SHF    3
#define HEADS  16
#define HD     32
#define MLPD   2048
#define SS     49
#define SP     64
#define NWIN   2048    /* B * 64 windows */
#define TOK    100352  /* B*H*W = NWIN*SS */

// ---------------- WMMA fragment helpers (per CDNA5 ISA 7.12.2 layouts) -------

__device__ __forceinline__ v16h ld_a(const h16* A, int lda, int row, int k0, int lane) {
    // A 16x32 f16: lanes 0-15 row M, elems 0-7 -> k=kb..kb+7, elems 8-15 -> k=kb+16..kb+23
    int kb = k0 + ((lane >> 4) << 3);
    const h16* base = A + (long)row * lda + kb;
    v8h lo = *(const v8h*)(base);
    v8h hi = *(const v8h*)(base + 16);
    return __builtin_shufflevector(lo, hi, 0,1,2,3,4,5,6,7,8,9,10,11,12,13,14,15);
}

__device__ __forceinline__ v16h ld_b(const h16* Bt, int ldb, int row, int k0, int lane) {
    // B 32x16 f16 from B^T [N,K] row-major: lanes 0-15 row N, elems 0-7 -> k=kb..kb+7, elems 8-15 -> k=kb+16..kb+23
    const h16* base = Bt + (long)row * ldb + k0 + ((lane >> 4) << 3);
    v8h lo = *(const v8h*)(base);
    v8h hi = *(const v8h*)(base + 16);
    return __builtin_shufflevector(lo, hi, 0,1,2,3,4,5,6,7,8,9,10,11,12,13,14,15);
}

__device__ __forceinline__ v8f wmma16(v16h a, v16h b, v8f c) {
    return __builtin_amdgcn_wmma_f32_16x16x32_f16(false, a, false, b, (short)0, c, false, false);
}

// ---------------- Tensor Data Mover: 2D tile global->LDS (ISA ch.8 D#) ------
// data_size = 2 bytes; tile_w elems per row, tile_h rows, row_stride in elems.

__device__ __forceinline__ void tdm_load_2d(unsigned lds_off, const void* gaddr,
                                            unsigned tile_w, unsigned tile_h,
                                            unsigned row_stride)
{
    unsigned long long ga = (unsigned long long)gaddr;
    v4u g0;
    g0[0] = 1u;                                            // count=1 (valid), user mode
    g0[1] = lds_off;                                       // lds_addr (bytes)
    g0[2] = (unsigned)(ga & 0xffffffffu);                  // global_addr[31:0]
    g0[3] = (unsigned)((ga >> 32) & 0x01ffffffu) | (2u << 30); // global_addr[56:32], type=2
    v8u g1;
    g1[0] = 1u << 16;                                      // wg_mask=0, data_size=1 (2B)
    g1[1] = (tile_w & 0xffffu) << 16;                      // tensor_dim0[15:0]
    g1[2] = (tile_w >> 16) | ((tile_h & 0xffffu) << 16);   // tensor_dim0[31:16], tensor_dim1[15:0]
    g1[3] = (tile_h >> 16) | ((tile_w & 0xffffu) << 16);   // tensor_dim1[31:16], tile_dim0
    g1[4] = tile_h & 0xffffu;                              // tile_dim1 (tile_dim2=0)
    g1[5] = row_stride;                                    // tensor_dim0_stride[31:0]
    g1[6] = 0u;                                            // stride[47:32], dim1_stride[15:0]
    g1[7] = 0u;
    asm volatile("tensor_load_to_lds %0, %1" :: "s"(g0), "s"(g1) : "memory");
}

// ---------------- weight transpose + fp32->f16 -------------------------------

__global__ void k_wt(const float* __restrict__ w, h16* __restrict__ wt, int K, int N) {
    long total = (long)K * N;
    for (long i = (long)blockIdx.x * blockDim.x + threadIdx.x; i < total;
         i += (long)gridDim.x * blockDim.x) {
        int k = (int)(i / N), n = (int)(i % N);
        wt[(long)n * K + k] = (h16)w[i];
    }
}

__global__ void k_zero_h(h16* __restrict__ p, long n) {
    for (long i = (long)blockIdx.x * blockDim.x + threadIdx.x; i < n;
         i += (long)gridDim.x * blockDim.x)
        p[i] = (h16)0.0f;
}

// ---------------- LayerNorm (opt. fused shift + window partition) ------------

template<int SHIFTED>
__global__ __launch_bounds__(256)
void k_ln(const float* __restrict__ x, const float* __restrict__ g,
          const float* __restrict__ bta, h16* __restrict__ out)
{
    int r = blockIdx.x;
    long src;
    if (SHIFTED) {
        int win = r / SS, s = r % SS;
        int b = win >> 6, wi = win & 63, wh = wi >> 3, ww = wi & 7;
        int si = s / WSZ, sj = s % WSZ;
        int hh = (wh * WSZ + si + SHF) % HH;
        int wc = (ww * WSZ + sj + SHF) % HH;
        src = ((long)b * (HH * HH) + hh * HH + wc) * DIMC;
    } else {
        src = (long)r * DIMC;
    }
    int t = threadIdx.x;
    float x0 = x[src + t], x1 = x[src + t + 256];
    __shared__ float s1[256], s2[256];
    s1[t] = x0 + x1;
    s2[t] = x0 * x0 + x1 * x1;
    __syncthreads();
    for (int off = 128; off > 0; off >>= 1) {
        if (t < off) { s1[t] += s1[t + off]; s2[t] += s2[t + off]; }
        __syncthreads();
    }
    float mean = s1[0] * (1.0f / (float)DIMC);
    float var  = s2[0] * (1.0f / (float)DIMC) - mean * mean;
    float rstd = rsqrtf(var + 1e-5f);
    out[(long)r * DIMC + t]       = (h16)((x0 - mean) * rstd * g[t]       + bta[t]);
    out[(long)r * DIMC + t + 256] = (h16)((x1 - mean) * rstd * g[t + 256] + bta[t + 256]);
}

// ---------------- generic WMMA GEMM: C = A[M,K] * Bt[N,K]^T + bias -----------
// A tile (64 rows x 32 k) staged via TDM into double-buffered LDS.
// MODE 0: qkv scatter (q scaled, K row-major, V transposed+padded)
// MODE 1: proj + window reverse + unshift + residual -> fp32
// MODE 2: GELU -> f16
// MODE 3: + residual -> fp32 out

template<int MODE>
__global__ __launch_bounds__(256)
void k_gemm(const h16* __restrict__ A, const h16* __restrict__ Bt,
            const float* __restrict__ bias, int M, int N, int K,
            h16* __restrict__ oQ, h16* __restrict__ oK, h16* __restrict__ oVt,
            h16* __restrict__ oH, float* __restrict__ oF,
            const float* __restrict__ resid)
{
    __shared__ __align__(16) h16 At[2][64 * 32];

    int lane = threadIdx.x & 31;
    int w    = threadIdx.x >> 5;
    int mblk = blockIdx.y * 64;
    int m0   = mblk + (w & 3) * 16;
    int n0   = blockIdx.x * 128 + (w >> 2) * 64;
    int nl   = lane & 15;
    int trow = (w & 3) * 16 + nl;         // A-tile-local fragment row
    bool issuer = (threadIdx.x < 32);

    unsigned lds0 = (unsigned)(unsigned long long)(void*)&At[0][0];
    unsigned lds1 = (unsigned)(unsigned long long)(void*)&At[1][0];

    if (issuer) {
        tdm_load_2d(lds0, A + (long)mblk * K, 32u, 64u, (unsigned)K);
        __builtin_amdgcn_s_wait_tensorcnt(0);
    }
    __syncthreads();

    v8f a0 = {}, a1 = {}, a2 = {}, a3 = {};
    for (int k0 = 0; k0 < K; k0 += 32) {
        int buf = (k0 >> 5) & 1;
        if (k0 + 32 < K && issuer)
            tdm_load_2d(buf ? lds0 : lds1, A + (long)mblk * K + (k0 + 32),
                        32u, 64u, (unsigned)K);

        const h16* Atile = &At[buf][0];
        v16h af = ld_a(Atile, 32, trow, 0, lane);
        if (k0 + 32 < K)
            __builtin_prefetch(Bt + (long)(n0 + nl) * K + k0 + 32, 0, 0);
        v16h b0 = ld_b(Bt, K, n0 +  0 + nl, k0, lane);
        v16h b1 = ld_b(Bt, K, n0 + 16 + nl, k0, lane);
        v16h b2 = ld_b(Bt, K, n0 + 32 + nl, k0, lane);
        v16h b3 = ld_b(Bt, K, n0 + 48 + nl, k0, lane);
        a0 = wmma16(af, b0, a0);
        a1 = wmma16(af, b1, a1);
        a2 = wmma16(af, b2, a2);
        a3 = wmma16(af, b3, a3);

        if (k0 + 32 < K) {
            if (issuer) __builtin_amdgcn_s_wait_tensorcnt(0);
            __syncthreads();
        }
    }

    int chalf = lane >> 4;
#pragma unroll
    for (int v = 0; v < 8; ++v) {
        int r = m0 + v + 8 * chalf;
        float vals[4] = { a0[v], a1[v], a2[v], a3[v] };

        int win = 0, s = 0;
        if (MODE == 0 || MODE == 1) { win = r / SS; s = r % SS; }

#pragma unroll
        for (int j = 0; j < 4; ++j) {
            int c = n0 + j * 16 + nl;
            float val = vals[j] + bias[c];
            if (MODE == 0) {
                int t  = c >> 9;          // 0=q 1=k 2=v
                int hd = (c >> 5) & 15;   // head
                int d  = c & 31;
                long rowqk = (((long)(win * HEADS + hd)) * SS + s) * HD + d;
                if (t == 0)      oQ[rowqk] = (h16)(val * 0.1767766953f);
                else if (t == 1) oK[rowqk] = (h16)val;
                else oVt[(((long)(win * HEADS + hd)) * HD + d) * SP + s] = (h16)val;
            } else if (MODE == 1) {
                int b  = win >> 6, wi = win & 63, wh = wi >> 3, ww = wi & 7;
                int si = s / WSZ, sj = s % WSZ;
                int hh = (wh * WSZ + si + SHF) % HH;
                int wc = (ww * WSZ + sj + SHF) % HH;
                long idx = ((long)b * (HH * HH) + hh * HH + wc) * DIMC + c;
                oF[idx] = resid[idx] + val;
            } else if (MODE == 2) {
                float gl = 0.5f * val * (1.0f + erff(val * 0.70710678f));
                oH[(long)r * N + c] = (h16)gl;
            } else {
                long idx = (long)r * N + c;
                oF[idx] = resid[idx] + val;
            }
        }
    }
}

// ---------------- windowed attention: one block per (window, head) -----------

__global__ __launch_bounds__(128)
void k_attn(const h16* __restrict__ Q, const h16* __restrict__ Kb,
            const h16* __restrict__ Vt, const float* __restrict__ rel_bias,
            h16* __restrict__ O)
{
    int blk  = blockIdx.x;           // win*HEADS + head
    int win  = blk >> 4;
    int head = blk & 15;
    int lane = threadIdx.x & 31;
    int wv   = threadIdx.x >> 5;     // wave = M-tile index (0..3)
    int nl   = lane & 15;

    const h16* q  = Q  + (long)blk * SS * HD;
    const h16* kk = Kb + (long)blk * SS * HD;
    const h16* vt = Vt + (long)blk * HD * SP;

    __shared__ __align__(16) float sc[64][68];
    __shared__ __align__(16) h16   P[64][64];

    int m0   = wv * 16;
    int arow = m0 + nl;

    // ---- scores = (Q*scale) @ K^T : K-dim = HD = 32 (one WMMA step) ----
    v16h af = ld_a(q, HD, arow, 0, lane);
    v16h b0 = ld_b(kk, HD,  0 + nl, 0, lane);
    v16h b1 = ld_b(kk, HD, 16 + nl, 0, lane);
    v16h b2 = ld_b(kk, HD, 32 + nl, 0, lane);
    v16h b3 = ld_b(kk, HD, 48 + nl, 0, lane);
    v8f s0 = {}, s1 = {}, s2 = {}, s3 = {};
    s0 = wmma16(af, b0, s0);
    s1 = wmma16(af, b1, s1);
    s2 = wmma16(af, b2, s2);
    s3 = wmma16(af, b3, s3);

    // ---- add rel-pos bias + shift mask, stash to LDS ----
    int wi = win & 63, wh = wi >> 3, ww = wi & 7;
    int chalf = lane >> 4;
#pragma unroll
    for (int v = 0; v < 8; ++v) {
        int sq = m0 + v + 8 * chalf;
        int qi = sq / WSZ, qj = sq % WSZ;
        int hq = wh * WSZ + qi, wq = ww * WSZ + qj;
        int cntq = ((hq < 49) ? 0 : ((hq < 53) ? 1 : 2)) * 3 +
                   ((wq < 49) ? 0 : ((wq < 53) ? 1 : 2));
        float vals[4] = { s0[v], s1[v], s2[v], s3[v] };
#pragma unroll
        for (int j = 0; j < 4; ++j) {
            int sk = j * 16 + nl;
            float val;
            if (sk < SS && sq < SS) {
                int ki = sk / WSZ, kj = sk % WSZ;
                int rpi = (qi - ki + 6) * 13 + (qj - kj + 6);
                int hk = wh * WSZ + ki, wk = ww * WSZ + kj;
                int cntk = ((hk < 49) ? 0 : ((hk < 53) ? 1 : 2)) * 3 +
                           ((wk < 49) ? 0 : ((wk < 53) ? 1 : 2));
                val = vals[j] + rel_bias[rpi * HEADS + head] +
                      ((cntq == cntk) ? 0.0f : -100.0f);
            } else {
                val = -1e30f;  // padded rows/cols
            }
            sc[sq][sk] = val;
        }
    }
    __syncthreads();

    // ---- softmax: one thread per row ----
    if (threadIdx.x < 64) {
        int row = threadIdx.x;
        float mx = -1e30f;
        for (int c = 0; c < 64; ++c) mx = fmaxf(mx, sc[row][c]);
        float sum = 0.0f;
        for (int c = 0; c < 64; ++c) {
            float e = __expf(sc[row][c] - mx);
            sc[row][c] = e;
            sum += e;
        }
        float inv = 1.0f / sum;
        for (int c = 0; c < 64; ++c) P[row][c] = (h16)(sc[row][c] * inv);
    }
    __syncthreads();

    // ---- out = P[64,64] @ V[64,32] : two WMMA K-steps, two N tiles ----
    v8f o0 = {}, o1 = {};
#pragma unroll
    for (int k0 = 0; k0 < 64; k0 += 32) {
        v16h pa  = ld_a(&P[0][0], 64, arow, k0, lane);
        v16h vb0 = ld_b(vt, SP,  0 + nl, k0, lane);
        v16h vb1 = ld_b(vt, SP, 16 + nl, k0, lane);
        o0 = wmma16(pa, vb0, o0);
        o1 = wmma16(pa, vb1, o1);
    }
#pragma unroll
    for (int v = 0; v < 8; ++v) {
        int sq = m0 + v + 8 * chalf;
        if (sq < SS) {
            long base = ((long)win * SS + sq) * DIMC + head * HD;
            O[base +      nl] = (h16)o0[v];
            O[base + 16 + nl] = (h16)o1[v];
        }
    }
}

// ---------------- host orchestration ----------------------------------------

extern "C" void kernel_launch(void* const* d_in, const int* in_sizes, int n_in,
                              void* d_out, int out_size, void* d_ws, size_t ws_size,
                              hipStream_t stream)
{
    const float* x        = (const float*)d_in[0];
    const float* qkv_w    = (const float*)d_in[1];
    const float* qkv_b    = (const float*)d_in[2];
    const float* proj_w   = (const float*)d_in[3];
    const float* proj_b   = (const float*)d_in[4];
    const float* rel_bias = (const float*)d_in[5];
    const float* n1g      = (const float*)d_in[6];
    const float* n1b      = (const float*)d_in[7];
    const float* n2g      = (const float*)d_in[8];
    const float* n2b      = (const float*)d_in[9];
    const float* w1       = (const float*)d_in[10];
    const float* b1       = (const float*)d_in[11];
    const float* w2       = (const float*)d_in[12];
    const float* b2       = (const float*)d_in[13];

    char* wsp = (char*)d_ws;
    auto alloc = [&](size_t bytes) -> char* {
        char* p = wsp;
        wsp += (bytes + 255) & ~(size_t)255;
        return p;
    };
    h16*   Xwin  = (h16*)  alloc((size_t)TOK * DIMC * 2);
    h16*   Qb    = (h16*)  alloc((size_t)TOK * DIMC * 2);
    h16*   Kb    = (h16*)  alloc((size_t)TOK * DIMC * 2);
    h16*   Vt    = (h16*)  alloc((size_t)NWIN * HEADS * HD * SP * 2);
    h16*   AO    = (h16*)  alloc((size_t)TOK * DIMC * 2);
    float* X1    = (float*)alloc((size_t)TOK * DIMC * 4);
    h16*   H2    = (h16*)  alloc((size_t)TOK * DIMC * 2);
    h16*   MID   = (h16*)  alloc((size_t)TOK * MLPD * 2);
    h16*   WqkvT = (h16*)  alloc((size_t)DIMC * 3 * DIMC * 2);
    h16*   WprT  = (h16*)  alloc((size_t)DIMC * DIMC * 2);
    h16*   W1T   = (h16*)  alloc((size_t)DIMC * MLPD * 2);
    h16*   W2T   = (h16*)  alloc((size_t)MLPD * DIMC * 2);
    (void)alloc(65536);  // slack for padded fragment reads past Q/K tails

    // weights -> f16 transposed [N,K]
    k_wt<<<1024, 256, 0, stream>>>(qkv_w,  WqkvT, DIMC, 3 * DIMC);
    k_wt<<<1024, 256, 0, stream>>>(proj_w, WprT,  DIMC, DIMC);
    k_wt<<<1024, 256, 0, stream>>>(w1,     W1T,   DIMC, MLPD);
    k_wt<<<1024, 256, 0, stream>>>(w2,     W2T,   MLPD, DIMC);
    k_zero_h<<<4096, 256, 0, stream>>>(Vt, (long)NWIN * HEADS * HD * SP);

    // LN1 + roll(-3,-3) + window partition
    k_ln<1><<<TOK, 256, 0, stream>>>(x, n1g, n1b, Xwin);

    // QKV projection
    k_gemm<0><<<dim3((3 * DIMC) / 128, TOK / 64), 256, 0, stream>>>(
        Xwin, WqkvT, qkv_b, TOK, 3 * DIMC, DIMC,
        Qb, Kb, Vt, nullptr, nullptr, nullptr);

    // windowed attention
    k_attn<<<NWIN * HEADS, 128, 0, stream>>>(Qb, Kb, Vt, rel_bias, AO);

    // output projection + window reverse + unshift + residual
    k_gemm<1><<<dim3(DIMC / 128, TOK / 64), 256, 0, stream>>>(
        AO, WprT, proj_b, TOK, DIMC, DIMC,
        nullptr, nullptr, nullptr, nullptr, X1, x);

    // LN2
    k_ln<0><<<TOK, 256, 0, stream>>>(X1, n2g, n2b, H2);

    // MLP fc1 + GELU
    k_gemm<2><<<dim3(MLPD / 128, TOK / 64), 256, 0, stream>>>(
        H2, W1T, b1, TOK, MLPD, DIMC,
        nullptr, nullptr, nullptr, MID, nullptr, nullptr);

    // MLP fc2 + residual -> output
    k_gemm<3><<<dim3(DIMC / 128, TOK / 64), 256, 0, stream>>>(
        MID, W2T, b2, TOK, DIMC, MLPD,
        nullptr, nullptr, nullptr, nullptr, (float*)d_out, X1);
}
